// GalerkinConv_44255343018777
// MI455X (gfx1250) — matrix operand order repaired and checked by reading, not compiled, not run on hardware
//
#include <hip/hip_runtime.h>

// ---------------------------------------------------------------------------
// Galerkin spectral conv for MI455X (gfx1250, wave32, WMMA).
//   B=16, C_IN=C_OUT=128, N=16384, MODES=128
// Memory-bound (~295 MB min traffic @ 23.3 TB/s ~ 13 us). bf16 WMMA (f32
// accum). LDS layouts keep k natural so every WMMA fragment is two contiguous
// 16B ds_load_b128; all f32->bf16 conversion uses native HW cvt; all LDS
// traffic is >= 8B vectorized except the 4B output staging stores.
// ---------------------------------------------------------------------------

typedef __attribute__((ext_vector_type(16))) __bf16        v16bf;
typedef __attribute__((ext_vector_type(4)))  __bf16        v4bf;
typedef __attribute__((ext_vector_type(8)))  float         v8f;
typedef __attribute__((ext_vector_type(4)))  float         v4f;
typedef __attribute__((ext_vector_type(2)))  unsigned int  v2u;
typedef __attribute__((ext_vector_type(4)))  unsigned int  v4u;

#define BATCH   16
#define CIN     128
#define COUT    128
#define NPTS    16384
#define MODES   128
#define MTOT    (BATCH * CIN)      // 2048 rows (b,c) stage1 / (b,o) stage3

// Fragment as two 16B LDS loads
union Frag { v4u q[2]; v16bf b; };

// native f32->bf16 (lowers to v_cvt_pk_bf16_f32 pairs)
__device__ __forceinline__ v2u cvt4(v4f v) {
  v4bf b = __builtin_convertvector(v, v4bf);
  return __builtin_bit_cast(v2u, b);
}

// A-fragment swizzle: within a 32-wide k block, element k is stored at
// pos = (k&7) | ((k&8)<<1) | ((k&16)>>1).  Then lane-half h reads its 16
// fragment values contiguously at [h*16 .. h*16+15] in ISA order.
__device__ __forceinline__ int aswz(int k) {
  return (k & 7) | ((k & 8) << 1) | ((k & 16) >> 1);
}

// ---------------------------------------------------------------------------
__global__ __launch_bounds__(256) void zero_f32(float* __restrict__ p, int n) {
  for (int i = blockIdx.x * blockDim.x + threadIdx.x; i < n;
       i += gridDim.x * blockDim.x)
    p[i] = 0.0f;
}

// ---------------------------------------------------------------------------
// Stage 1: xco[2048,128] += x[2048,16384] * wbases[16384,128]   (split-K=32)
// ---------------------------------------------------------------------------
__global__ __launch_bounds__(256) void stage1_xco(const float* __restrict__ x,
                                                  const float* __restrict__ wb,
                                                  float* __restrict__ xco) {
  __shared__ __align__(16) unsigned short As[64][48];    // [m][k natural]
  __shared__ __align__(16) unsigned short Bt[128][48];   // [n][k natural]

  const int t    = threadIdx.x;
  const int lane = t & 31, wave = t >> 5;
  const int half = lane >> 4, lm = lane & 15;
  const int m0    = blockIdx.x * 64;       // 32 M-tiles
  const int kBase = blockIdx.y * 512;      // 32 K-chunks

  const int rm = (wave & 3) * 16;
  const int nb = (wave >> 2) * 64;

  // A-loader coords: one 16B natural-order 8-chunk per thread
  const int ar = t >> 2;                   // A row 0..63
  const int ac = (t & 3) * 8;              // A col {0,8,16,24}
  const int ap = ac;
  // B-loader coords: one 4x4 register transpose per thread
  const int kb = (t >> 5) * 4;             // k row block {0,4,..,28}
  const int nc = (t & 31) * 4;             // n col {0,4,..,124}

  v8f acc[4] = {};

  for (int it = 0; it < 16; ++it) {
    const int k0 = kBase + it * 32;

    // ---- A tile 64x32: coalesced 16B loads, one 16B LDS store
    {
      const float* src = x + (size_t)(m0 + ar) * NPTS + k0 + ac;
      v4f a0 = reinterpret_cast<const v4f*>(src)[0];
      v4f a1 = reinterpret_cast<const v4f*>(src)[1];
      v2u l0 = cvt4(a0), l1 = cvt4(a1);
      v4u pk = {l0[0], l0[1], l1[0], l1[1]};
      *reinterpret_cast<v4u*>(&As[ar][ap]) = pk;
      if (it + 1 < 16) __builtin_prefetch(src + 32, 0, 3);   // next k tile
    }
    // ---- B tile 32k x 128n -> [n][k]: 4x4 register transpose, 8B LDS stores
    {
      const float* wrow = wb + (size_t)(k0 + kb) * MODES + nc;
      v4f r0 = *reinterpret_cast<const v4f*>(wrow);
      v4f r1 = *reinterpret_cast<const v4f*>(wrow + MODES);
      v4f r2 = *reinterpret_cast<const v4f*>(wrow + 2 * MODES);
      v4f r3 = *reinterpret_cast<const v4f*>(wrow + 3 * MODES);
#pragma unroll
      for (int q = 0; q < 4; ++q) {
        v4f col = {r0[q], r1[q], r2[q], r3[q]};
        *reinterpret_cast<v2u*>(&Bt[nc + q][kb]) = cvt4(col);
      }
    }
    __syncthreads();

    Frag a;
    {
      const v4u* pa = reinterpret_cast<const v4u*>(&As[rm + lm][half * 16]);
      a.q[0] = pa[0]; a.q[1] = pa[1];
    }
#pragma unroll
    for (int nt = 0; nt < 4; ++nt) {
      Frag b;
      const v4u* pb = reinterpret_cast<const v4u*>(&Bt[nb + nt * 16 + lm][half * 16]);
      b.q[0] = pb[0]; b.q[1] = pb[1];
      acc[nt] = __builtin_amdgcn_wmma_f32_16x16x32_bf16(
          false, a.b, false, b.b, (short)0, acc[nt], false, false);
    }
    __syncthreads();
  }

  // split-K reduction into L2-resident xco (native global_atomic_add_f32)
#pragma unroll
  for (int nt = 0; nt < 4; ++nt) {
    const int col = nb + nt * 16 + lm;
#pragma unroll
    for (int v = 0; v < 8; ++v) {
      const int row = m0 + rm + v + half * 8;
      atomicAdd(&xco[row * MODES + col], acc[nt][v]);
    }
  }
}

// ---------------------------------------------------------------------------
// Stage 2: per-mode mixing  xhat[b,o,k] = sum_i xco[b,i,k] * W[i,o,k]
// ---------------------------------------------------------------------------
__global__ __launch_bounds__(256) void stage2_mix(const float* __restrict__ xco,
                                                  const float* __restrict__ W,
                                                  float* __restrict__ xhat) {
  __shared__ __align__(16) unsigned short As[16][136];    // [b][i natural]
  __shared__ __align__(16) unsigned short Bt[128][136];   // [o][i natural]

  const int t    = threadIdx.x;
  const int lane = t & 31, wave = t >> 5;
  const int half = lane >> 4, lm = lane & 15;
  const int k    = blockIdx.x;              // mode

  // gather xco[:, :, k] (stride MODES), 4-at-a-time -> one 8B LDS store
#pragma unroll
  for (int j = 0; j < 2; ++j) {
    const int f  = t * 2 + j;               // 0..511 blocks of 4
    const int b  = f >> 5;
    const int i4 = (f & 31) * 4;
    v4f col;
#pragma unroll
    for (int q = 0; q < 4; ++q) col[q] = xco[(b * CIN + i4 + q) * MODES + k];
    *reinterpret_cast<v2u*>(&As[b][i4]) = cvt4(col);
  }
  // gather W[:, :, k] -> [o][i], 4-at-a-time -> one 8B LDS store
#pragma unroll
  for (int j = 0; j < 16; ++j) {
    const int f  = t * 16 + j;              // 0..4095 blocks of 4
    const int o  = f & 127;
    const int i4 = (f >> 7) * 4;
    v4f col;
#pragma unroll
    for (int q = 0; q < 4; ++q) col[q] = W[((size_t)(i4 + q) * COUT + o) * MODES + k];
    *reinterpret_cast<v2u*>(&Bt[o][i4]) = cvt4(col);
  }
  __syncthreads();

  const int o0 = wave * 16;
  v8f acc = {};
#pragma unroll
  for (int kk = 0; kk < 128; kk += 32) {    // reduce over i
    Frag a, b;
    const v4u* pa = reinterpret_cast<const v4u*>(&As[lm][kk + half * 16]);
    a.q[0] = pa[0]; a.q[1] = pa[1];
    const v4u* pb = reinterpret_cast<const v4u*>(&Bt[o0 + lm][kk + half * 16]);
    b.q[0] = pb[0]; b.q[1] = pb[1];
    acc = __builtin_amdgcn_wmma_f32_16x16x32_bf16(
        false, a.b, false, b.b, (short)0, acc, false, false);
  }

#pragma unroll
  for (int v = 0; v < 8; ++v) {
    const int b = v + half * 8;
    const int o = o0 + lm;
    xhat[(b * COUT + o) * MODES + k] = acc[v];
  }
}

// ---------------------------------------------------------------------------
// Stage 3: out[2048,16384] = xhat[2048,128] * bases^T[128,16384]
// block tile 64(M) x 128(N), full K=128 in LDS; output staged in LDS for
// fully coalesced 16B stores (dominant 134MB HBM stream).
// ---------------------------------------------------------------------------
struct S3Smem {
  union {
    struct {
      unsigned short As[64][136];    // [m][k natural]
      unsigned short Bs[128][136];   // [x][k natural]  B(k,x)=bases[x][k]
    } in;
    float Cs[64][132];               // output staging (overlaid)
  };
};

__global__ __launch_bounds__(256) void stage3_out(const float* __restrict__ xhat,
                                                  const float* __restrict__ bases,
                                                  float* __restrict__ out) {
  __shared__ __align__(16) S3Smem sm;

  const int t    = threadIdx.x;
  const int lane = t & 31, wave = t >> 5;
  const int half = lane >> 4, lm = lane & 15;
  const int n0 = blockIdx.x * 128;          // 128 N-tiles over x
  const int m0 = blockIdx.y * 64;           // 32 M-tiles

  // A tile 64x128 (16B coalesced loads, natural k order, 8B LDS stores)
#pragma unroll
  for (int j = 0; j < 8; ++j) {
    const int f  = t + 256 * j;
    const int r  = f >> 5;
    const int c4 = (f & 31) * 4;
    const int p  = c4;   // 4-run stays contiguous
    v4f v = *reinterpret_cast<const v4f*>(xhat + (size_t)(m0 + r) * MODES + c4);
    *reinterpret_cast<v2u*>(&sm.in.As[r][p]) = cvt4(v);
  }
  // B tile 128x128 (bases rows already [x][k])
#pragma unroll
  for (int j = 0; j < 16; ++j) {
    const int f  = t + 256 * j;
    const int r  = f >> 5;
    const int c4 = (f & 31) * 4;
    v4f v = *reinterpret_cast<const v4f*>(bases + (size_t)(n0 + r) * MODES + c4);
    *reinterpret_cast<v2u*>(&sm.in.Bs[r][c4]) = cvt4(v);
  }
  __syncthreads();

  const int rm = (wave & 3) * 16;
  const int nb = (wave >> 2) * 64;
  v8f acc[4] = {};

#pragma unroll
  for (int kk = 0; kk < 128; kk += 32) {
    Frag a;
    const v4u* pa = reinterpret_cast<const v4u*>(&sm.in.As[rm + lm][kk + half * 16]);
    a.q[0] = pa[0]; a.q[1] = pa[1];
#pragma unroll
    for (int nt = 0; nt < 4; ++nt) {
      Frag b;
      const v4u* pb =
          reinterpret_cast<const v4u*>(&sm.in.Bs[nb + nt * 16 + lm][kk + half * 16]);
      b.q[0] = pb[0]; b.q[1] = pb[1];
      acc[nt] = __builtin_amdgcn_wmma_f32_16x16x32_bf16(
          false, a.b, false, b.b, (short)0, acc[nt], false, false);
    }
  }

  // stage output tile in LDS (overlaid on A/B space), then coalesced stores
  __syncthreads();
#pragma unroll
  for (int nt = 0; nt < 4; ++nt)
#pragma unroll
    for (int v = 0; v < 8; ++v)
      sm.Cs[rm + v + half * 8][nb + nt * 16 + lm] = acc[nt][v];
  __syncthreads();

#pragma unroll
  for (int j = 0; j < 8; ++j) {
    const int f  = t + 256 * j;
    const int r  = f >> 5;
    const int c4 = (f & 31) * 4;
    v4f v = *reinterpret_cast<const v4f*>(&sm.Cs[r][c4]);
    *reinterpret_cast<v4f*>(out + (size_t)(m0 + r) * NPTS + n0 + c4) = v;
  }
}

// ---------------------------------------------------------------------------
extern "C" void kernel_launch(void* const* d_in, const int* in_sizes, int n_in,
                              void* d_out, int out_size, void* d_ws, size_t ws_size,
                              hipStream_t stream) {
  const float* x      = (const float*)d_in[0];   // [16,128,16384]
  const float* wbases = (const float*)d_in[1];   // [16384,128]
  const float* bases  = (const float*)d_in[2];   // [16384,128]
  const float* W      = (const float*)d_in[3];   // [128,128,128]
  float*       out    = (float*)d_out;           // [16,128,16384]

  float* xco  = (float*)d_ws;                    // [2048,128] f32 (1 MB)
  float* xhat = xco + (size_t)MTOT * MODES;      // [2048,128] f32 (1 MB)

  zero_f32<<<256, 256, 0, stream>>>(xco, MTOT * MODES);
  stage1_xco<<<dim3(32, 32), 256, 0, stream>>>(x, wbases, xco);
  stage2_mix<<<MODES, 256, 0, stream>>>(xco, W, xhat);
  stage3_out<<<dim3(128, 32), 256, 0, stream>>>(xhat, bases, out);
}
